// FrameNetProtein01_44117904065156
// MI455X (gfx1250) — compile-verified, matrix operand order checked by reading, not run on hardware
//
#include <hip/hip_runtime.h>

#define N_RES     16384
#define EMB       128
#define NUM_RADIAL 64
#define NUM_EDGES 786432
#define NODES     (3 * N_RES)
#define CUTOFF    5.0f
#define GAMMA     10.0f

typedef __attribute__((ext_vector_type(16))) _Float16 v16h;
typedef __attribute__((ext_vector_type(8)))  float    v8f;

#define V8F_ZERO {0.f,0.f,0.f,0.f,0.f,0.f,0.f,0.f}

// ---------------------------------------------------------------------------
// WMMA helpers (gfx1250 wave32).  Assumed fragment layouts per CDNA5 ISA:
//   A (f16 16x32): lane l holds row M=l&15; half h -> K = 8*(l>>4) + (h<8?h:h+8)
//   B (f16 32x16): lane l holds col N=l&15; same K mapping (symmetric)
//   C/D (f32 16x16): vgpr r, lane l -> M = r + 8*(l>>4), N = l&15
// ---------------------------------------------------------------------------
__device__ __forceinline__ v8f wmma16(v16h a, v16h b, v8f c) {
  return __builtin_amdgcn_wmma_f32_16x16x32_f16(false, a, false, b, (short)0, c,
                                                false, false);
}

__device__ __forceinline__ int frag_k(int lane, int h) {
  return ((lane >> 4) << 3) + (h < 8 ? h : h + 8);
}

// Build an A-fragment of Gaussian RBF features directly in registers.
__device__ __forceinline__ v16h rbf_frag(float d, int lane, int kbase) {
  v16h a;
#pragma unroll
  for (int h = 0; h < 16; ++h) {
    int K = kbase + frag_k(lane, h);
    float mu = (float)K * (CUTOFF / 63.0f);   // linspace(0, 5, 64)
    float t = d - mu;
    a[h] = (_Float16)__expf(-GAMMA * t * t);
  }
  return a;
}

__device__ __forceinline__ void load_pos(int node, const float* pN,
                                         const float* pCa, const float* pC,
                                         float& x, float& y, float& z, int& a) {
  int res = node / 3;
  a = node - res * 3;
  const float* p = (a == 0) ? pN : ((a == 1) ? pCa : pC);
  x = p[res * 3 + 0]; y = p[res * 3 + 1]; z = p[res * 3 + 2];
}

// ---------------------------------------------------------------------------
// Prep kernels
// ---------------------------------------------------------------------------
__global__ __launch_bounds__(256) void k_zero(float4* p, int n4) {
  for (int i = blockIdx.x * blockDim.x + threadIdx.x; i < n4;
       i += gridDim.x * blockDim.x)
    p[i] = make_float4(0.f, 0.f, 0.f, 0.f);
}

// Permute a row-major [Ktot,Ntot] f32 weight into fragment-major f16:
// dst half index = ((t*(Ktot/32)+s)*32 + lane)*16 + h, value = src[K*Ntot+N].
__global__ __launch_bounds__(256) void k_pack(const float* __restrict__ src,
                                              _Float16* __restrict__ dst,
                                              int ksShift, int Ntot, int total) {
  int ksMask = (1 << ksShift) - 1;
  for (int idx = blockIdx.x * blockDim.x + threadIdx.x; idx < total;
       idx += gridDim.x * blockDim.x) {
    int frag = idx >> 9;
    int rem = idx & 511;
    int l = rem >> 4, h = rem & 15;
    int t = frag >> ksShift, s = frag & ksMask;
    int N = t * 16 + (l & 15);
    int K = 32 * s + ((l >> 4) << 3) + (h < 8 ? h : h + 8);
    dst[idx] = (_Float16)src[K * Ntot + N];
  }
}

__global__ __launch_bounds__(256) void k_outinit(const int* __restrict__ rtype,
                                                 const float* __restrict__ remb,
                                                 float* __restrict__ out) {
  for (int idx = blockIdx.x * blockDim.x + threadIdx.x; idx < N_RES * EMB;
       idx += gridDim.x * blockDim.x) {
    int res = idx >> 7, f = idx & 127;
    out[idx] = remb[rtype[res] * EMB + f];
  }
}

// ---------------------------------------------------------------------------
// Edge kernel: 8 waves/block, 32 edges/wave (2 subtiles of 16).
// radial[16,128] = rbf(dist)[16,64] @ W[64,128] via WMMA; each B fragment is
// reused across both subtiles.  Scatter: lig[i] += x[i%3] + x[j%3]*radial.
// Per-edge index math is precomputed once at setup:
//   p1 = i*EMB (atomic base offset), p2 = (i%3)*EMB | ((j%3)*EMB)<<16
// so the inner loop is and/shr + 2 LDS loads + fma + global atomic.
// ---------------------------------------------------------------------------
__global__ __launch_bounds__(256) void k_edges(
    const int* __restrict__ eidx, const float* __restrict__ pN,
    const float* __restrict__ pCa, const float* __restrict__ pC,
    const float* __restrict__ bbemb, const float* __restrict__ ligb,
    const _Float16* __restrict__ WligF, float* __restrict__ lig) {
  __shared__ float s_bb[3 * EMB];
  __shared__ float s_lb[EMB];
  __shared__ float s_dist[8][32];
  __shared__ int s_p1[8][32];
  __shared__ int s_p2[8][32];

  for (int t = threadIdx.x; t < 3 * EMB; t += 256) s_bb[t] = bbemb[t];
  for (int t = threadIdx.x; t < EMB; t += 256) s_lb[t] = ligb[t];
  __syncthreads();

  int lane = threadIdx.x & 31;
  int wv = threadIdx.x >> 5;
  int base = (blockIdx.x * 8 + wv) * 32;
  int nlo = lane & 15;
  int mbase = (lane >> 4) << 3;

  // setup: one edge per lane (EXEC stays all-ones)
  {
    int e = base + lane;
    int i = eidx[e];
    int j = eidx[NUM_EDGES + e];
    float ix, iy, iz, jx, jy, jz;
    int i3, j3;
    load_pos(i, pN, pCa, pC, ix, iy, iz, i3);
    load_pos(j, pN, pCa, pC, jx, jy, jz, j3);
    float dx = ix - jx, dy = iy - jy, dz = iz - jz;
    s_dist[wv][lane] = sqrtf(dx * dx + dy * dy + dz * dz);
    s_p1[wv][lane] = i * EMB;
    s_p2[wv][lane] = (i3 * EMB) | ((j3 * EMB) << 16);
  }
  __builtin_amdgcn_wave_barrier();  // wave-local LDS handoff (in-order DS pipe)

  // A fragments for the 2 subtiles (held in registers across the t loop)
  v16h a0[2], a1[2];
#pragma unroll
  for (int st = 0; st < 2; ++st) {
    float d = s_dist[wv][st * 16 + nlo];
    a0[st] = rbf_frag(d, lane, 0);
    a1[st] = rbf_frag(d, lane, 32);
  }
  // packed index words for the 8 rows this lane owns in each subtile
  int w1[2][8], w2[2][8];
#pragma unroll
  for (int st = 0; st < 2; ++st)
#pragma unroll
    for (int r = 0; r < 8; ++r) {
      w1[st][r] = s_p1[wv][st * 16 + mbase + r];
      w2[st][r] = s_p2[wv][st * 16 + mbase + r];
    }

  const v16h* Bf = (const v16h*)WligF;
#pragma unroll 1
  for (int t = 0; t < 8; ++t) {
    v16h b0 = Bf[(t * 2 + 0) * 32 + lane];
    v16h b1 = Bf[(t * 2 + 1) * 32 + lane];
    int ch = t * 16 + nlo;
    float bias = s_lb[ch];
#pragma unroll
    for (int st = 0; st < 2; ++st) {
      v8f acc = V8F_ZERO;
      acc = wmma16(a0[st], b0, acc);
      acc = wmma16(a1[st], b1, acc);
#pragma unroll
      for (int r = 0; r < 8; ++r) {
        int p2 = w2[st][r];
        float xi = s_bb[(p2 & 0xFFFF) + ch];
        float xj = s_bb[(p2 >> 16) + ch];
        float radial = acc[r] + bias;
        unsafeAtomicAdd(&lig[w1[st][r] + ch], xi + xj * radial);
      }
    }
  }
}

// ---------------------------------------------------------------------------
// Residue kernel: 4 waves/block, 16 residues/wave.  Chained WMMA GEMMs:
// radial -> er@e1 -> scalarization MLP -> s@e2a(relu) -> @e2b, out += 0.5*s.
// ---------------------------------------------------------------------------
#define K3W 4
__global__ __launch_bounds__(128) void k_res(
    const float* __restrict__ pN, const float* __restrict__ pCa,
    const float* __restrict__ pC, const float* __restrict__ sc1W,
    const float* __restrict__ sc1b, const float* __restrict__ sc2W,
    const float* __restrict__ sc2b, const float* __restrict__ bbb,
    const float* __restrict__ e1b, const float* __restrict__ e2ab,
    const float* __restrict__ e2bb, const _Float16* __restrict__ WbbF,
    const _Float16* __restrict__ We1F, const _Float16* __restrict__ We2aF,
    const _Float16* __restrict__ We2bF, const float* __restrict__ lig,
    float* __restrict__ out) {
  __shared__ float s_sc1W[96], s_sc1b[32], s_sc2W[32], s_sc2b;
  __shared__ float s_bbb[128], s_e1b[128], s_e2ab[256], s_e2bb[128];
  __shared__ float s_geo[K3W][16 * 17];          // d1,d2,vec1[3],vec2[3],frame[9]
  __shared__ _Float16 s_ruv[K3W][16 * EMB];
  __shared__ _Float16 s_sv[K3W][16 * EMB];
  __shared__ _Float16 s_hv[K3W][16 * EMB];

  for (int t = threadIdx.x; t < 96; t += 128) s_sc1W[t] = sc1W[t];
  for (int t = threadIdx.x; t < 32; t += 128) { s_sc1b[t] = sc1b[t]; s_sc2W[t] = sc2W[t]; }
  if (threadIdx.x == 0) s_sc2b = sc2b[0];
  for (int t = threadIdx.x; t < 128; t += 128) {
    s_bbb[t] = bbb[t]; s_e1b[t] = e1b[t]; s_e2bb[t] = e2bb[t];
  }
  for (int t = threadIdx.x; t < 256; t += 128) s_e2ab[t] = e2ab[t];
  __syncthreads();

  int lane = threadIdx.x & 31;
  int wv = threadIdx.x >> 5;
  int resBase = (blockIdx.x * K3W + wv) * 16;
  float* g = s_geo[wv];
  _Float16* ruv = s_ruv[wv];
  _Float16* sbuf = s_sv[wv];
  _Float16* hbuf = s_hv[wv];
  int nlo = lane & 15;
  int mbase = (lane >> 4) << 3;

  if (lane < 16) {
    int rg = resBase + lane;
    float nx = pN[rg*3+0], ny = pN[rg*3+1], nz = pN[rg*3+2];
    float ax = pCa[rg*3+0], ay = pCa[rg*3+1], az = pCa[rg*3+2];
    float cx = pC[rg*3+0], cy = pC[rg*3+1], cz = pC[rg*3+2];
    float v1x = ax-nx, v1y = ay-ny, v1z = az-nz;
    float d1 = sqrtf(v1x*v1x + v1y*v1y + v1z*v1z);
    float i1 = 1.0f / fmaxf(d1, 1e-8f);
    float f1x = v1x*i1, f1y = v1y*i1, f1z = v1z*i1;
    float v2x = cx-ax, v2y = cy-ay, v2z = cz-az;
    float d2 = sqrtf(v2x*v2x + v2y*v2y + v2z*v2z);
    float i2 = 1.0f / fmaxf(d2, 1e-8f);
    float f2x = v2x*i2, f2y = v2y*i2, f2z = v2z*i2;
    float crx = f1y*f2z - f1z*f2y;
    float cry = f1z*f2x - f1x*f2z;
    float crz = f1x*f2y - f1y*f2x;
    float dc = sqrtf(crx*crx + cry*cry + crz*crz);
    float ic = 1.0f / fmaxf(dc, 1e-8f);
    float fcx = crx*ic, fcy = cry*ic, fcz = crz*ic;
    float* gr = &g[lane * 17];
    gr[0] = d1; gr[1] = d2;
    gr[2] = v1x; gr[3] = v1y; gr[4] = v1z;
    gr[5] = v2x; gr[6] = v2y; gr[7] = v2z;
    gr[8+0] = f1x; gr[8+1] = f2x; gr[8+2] = fcx;   // frame[d=0][k]
    gr[8+3] = f1y; gr[8+4] = f2y; gr[8+5] = fcy;   // frame[d=1][k]
    gr[8+6] = f1z; gr[8+7] = f2z; gr[8+8] = fcz;   // frame[d=2][k]
  }
  __builtin_amdgcn_wave_barrier();

  const v16h* Bbb = (const v16h*)WbbF;
  const v16h* Be1 = (const v16h*)We1F;
  const v16h* Be2a = (const v16h*)We2aF;
  const v16h* Be2b = (const v16h*)We2bF;

  for (int kind = 0; kind < 2; ++kind) {
    // ---- radial = rbf(dist) @ bb_rbf_W + b;  ruv = radial*u*v (f16) ----
    float d = g[nlo * 17 + kind];
    v16h a0 = rbf_frag(d, lane, 0);
    v16h a1 = rbf_frag(d, lane, 32);
#pragma unroll
    for (int t = 0; t < 8; ++t) {
      v8f acc = V8F_ZERO;
      acc = wmma16(a0, Bbb[(t * 2 + 0) * 32 + lane], acc);
      acc = wmma16(a1, Bbb[(t * 2 + 1) * 32 + lane], acc);
      int ch = t * 16 + nlo;
      float bias = s_bbb[ch];
#pragma unroll
      for (int r = 0; r < 8; ++r) {
        int rm = mbase + r;
        int node = 3 * (resBase + rm) + kind;
        float u = lig[node * EMB + ch];
        float v = lig[(node + 1) * EMB + ch];
        ruv[rm * EMB + ch] = (_Float16)((acc[r] + bias) * u * v);
      }
    }
    __builtin_amdgcn_wave_barrier();

    // ---- er_d = (vec_d*ruv) @ e1_W + e1_b; sc projection; tiny MLP -> s ----
    _Float16 vh0 = (_Float16)g[nlo * 17 + 2 + kind * 3 + 0];
    _Float16 vh1 = (_Float16)g[nlo * 17 + 2 + kind * 3 + 1];
    _Float16 vh2 = (_Float16)g[nlo * 17 + 2 + kind * 3 + 2];
#pragma unroll
    for (int t = 0; t < 8; ++t) {
      v8f acc0 = V8F_ZERO, acc1 = V8F_ZERO, acc2 = V8F_ZERO;
#pragma unroll
      for (int s = 0; s < 4; ++s) {
        v16h rf;
#pragma unroll
        for (int h = 0; h < 16; ++h)
          rf[h] = ruv[nlo * EMB + 32 * s + frag_k(lane, h)];
        v16h b = Be1[(t * 4 + s) * 32 + lane];
        acc0 = wmma16(rf * vh0, b, acc0);
        acc1 = wmma16(rf * vh1, b, acc1);
        acc2 = wmma16(rf * vh2, b, acc2);
      }
      int f = t * 16 + nlo;
      float b1 = s_e1b[f];
#pragma unroll
      for (int r = 0; r < 8; ++r) {
        int rm = mbase + r;
        const float* fr = &g[rm * 17 + 8];
        float er0 = acc0[r] + b1;
        float er1 = acc1[r] + b1;
        float er2 = acc2[r] + b1;
        float sc0 = er0 * fr[0] + er1 * fr[3] + er2 * fr[6];
        float sc1 = er0 * fr[1] + er1 * fr[4] + er2 * fr[7];
        float sc2v = er0 * fr[2] + er1 * fr[5] + er2 * fr[8];
        float s1 = s_sc2b;
#pragma unroll 8
        for (int c = 0; c < 32; ++c) {
          float z = sc0 * s_sc1W[c] + sc1 * s_sc1W[32 + c] +
                    sc2v * s_sc1W[64 + c] + s_sc1b[c];
          float sig = 1.0f / (1.0f + __expf(-z));
          s1 += z * sig * s_sc2W[c];
        }
        sbuf[rm * EMB + f] = (_Float16)(s1 + sc0);
      }
    }
    __builtin_amdgcn_wave_barrier();

    // ---- s @ e2a (relu, 256) then @ e2b (128); out += 0.5*(.. + bias) ----
    v16h aS[4];
#pragma unroll
    for (int s = 0; s < 4; ++s) {
#pragma unroll
      for (int h = 0; h < 16; ++h)
        aS[s][h] = sbuf[nlo * EMB + 32 * s + frag_k(lane, h)];
    }
#pragma unroll
    for (int kh = 0; kh < 2; ++kh) {      // 256 hidden channels in two halves
#pragma unroll
      for (int t2 = 0; t2 < 8; ++t2) {
        int tt = kh * 8 + t2;
        v8f acc = V8F_ZERO;
#pragma unroll
        for (int s = 0; s < 4; ++s)
          acc = wmma16(aS[s], Be2a[(tt * 4 + s) * 32 + lane], acc);
        int ch = tt * 16 + nlo;
        float bias = s_e2ab[ch];
#pragma unroll
        for (int r = 0; r < 8; ++r) {
          int rm = mbase + r;
          float v = acc[r] + bias;
          hbuf[rm * EMB + t2 * 16 + nlo] = (_Float16)(v > 0.0f ? v : 0.0f);
        }
      }
      __builtin_amdgcn_wave_barrier();
#pragma unroll
      for (int t3 = 0; t3 < 8; ++t3) {
        v8f acc = V8F_ZERO;
#pragma unroll
        for (int s = 0; s < 4; ++s) {
          v16h aH;
#pragma unroll
          for (int h = 0; h < 16; ++h)
            aH[h] = hbuf[nlo * EMB + 32 * s + frag_k(lane, h)];
          acc = wmma16(aH, Be2b[(t3 * 8 + kh * 4 + s) * 32 + lane], acc);
        }
        int f = t3 * 16 + nlo;
#pragma unroll
        for (int r = 0; r < 8; ++r) {
          int rm = mbase + r;
          float add = 0.5f * acc[r] + (kh == 0 ? 0.5f * s_e2bb[f] : 0.0f);
          float* o = &out[(resBase + rm) * EMB + f];
          *o = *o + add;   // element owned by this lane; plain RMW
        }
      }
      __builtin_amdgcn_wave_barrier();
    }
  }
}

// ---------------------------------------------------------------------------
extern "C" void kernel_launch(void* const* d_in, const int* in_sizes, int n_in,
                              void* d_out, int out_size, void* d_ws,
                              size_t ws_size, hipStream_t stream) {
  const float* pos_N        = (const float*)d_in[0];
  const float* pos_Ca       = (const float*)d_in[1];
  const float* pos_C        = (const float*)d_in[2];
  const int*   residue_type = (const int*)d_in[3];
  const int*   edge_index   = (const int*)d_in[5];
  const float* residue_emb  = (const float*)d_in[6];
  const float* backbone_emb = (const float*)d_in[7];
  const float* lig_rbf_W    = (const float*)d_in[8];
  const float* lig_rbf_b    = (const float*)d_in[9];
  const float* bb_rbf_W     = (const float*)d_in[10];
  const float* bb_rbf_b     = (const float*)d_in[11];
  const float* e1_W         = (const float*)d_in[12];
  const float* e1_b         = (const float*)d_in[13];
  const float* e2a_W        = (const float*)d_in[14];
  const float* e2a_b        = (const float*)d_in[15];
  const float* e2b_W        = (const float*)d_in[16];
  const float* e2b_b        = (const float*)d_in[17];
  const float* sc1_W        = (const float*)d_in[18];
  const float* sc1_b        = (const float*)d_in[19];
  const float* sc2_W        = (const float*)d_in[20];
  const float* sc2_b        = (const float*)d_in[21];
  float* out = (float*)d_out;

  char* ws = (char*)d_ws;
  float* lig = (float*)ws;
  size_t off = (size_t)NODES * EMB * sizeof(float);
  _Float16* WligF = (_Float16*)(ws + off); off += (size_t)64 * 128 * 2;
  _Float16* WbbF  = (_Float16*)(ws + off); off += (size_t)64 * 128 * 2;
  _Float16* We1F  = (_Float16*)(ws + off); off += (size_t)128 * 128 * 2;
  _Float16* We2aF = (_Float16*)(ws + off); off += (size_t)128 * 256 * 2;
  _Float16* We2bF = (_Float16*)(ws + off); off += (size_t)256 * 128 * 2;

  k_zero<<<2048, 256, 0, stream>>>((float4*)lig, NODES * EMB / 4);
  k_pack<<<32, 256, 0, stream>>>(lig_rbf_W, WligF, 1, 128, 64 * 128);
  k_pack<<<32, 256, 0, stream>>>(bb_rbf_W, WbbF, 1, 128, 64 * 128);
  k_pack<<<64, 256, 0, stream>>>(e1_W, We1F, 2, 128, 128 * 128);
  k_pack<<<128, 256, 0, stream>>>(e2a_W, We2aF, 2, 256, 128 * 256);
  k_pack<<<128, 256, 0, stream>>>(e2b_W, We2bF, 3, 128, 256 * 128);
  k_outinit<<<2048, 256, 0, stream>>>(residue_type, residue_emb, out);
  k_edges<<<NUM_EDGES / 256, 256, 0, stream>>>(edge_index, pos_N, pos_Ca, pos_C,
                                               backbone_emb, lig_rbf_b, WligF,
                                               lig);
  k_res<<<N_RES / 64, 128, 0, stream>>>(pos_N, pos_Ca, pos_C, sc1_W, sc1_b,
                                        sc2_W, sc2_b, bb_rbf_b, e1_b, e2a_b,
                                        e2b_b, WbbF, We1F, We2aF, We2bF, lig,
                                        out);
  (void)in_sizes; (void)n_in; (void)out_size; (void)ws_size;
}
